// MutagNet_17643725652192
// MI455X (gfx1250) — matrix operand order repaired and checked by reading, not compiled, not run on hardware
//
#include <hip/hip_runtime.h>

// MutagNet (GINE, 2 layers) for gfx1250 / MI455X.
// Dense MLPs via v_wmma_f32_16x16x32_f16 (K=32 == feature width).
// Edge embedding recomputed inline in the scatter kernel (saves 256MB/layer HBM).
// h/agg (25.6MB each) are L2-resident on the 192MB L2 -> scatter is L2-bound.

typedef __attribute__((ext_vector_type(16))) _Float16 v16h;
typedef __attribute__((ext_vector_type(8)))  float    v8f;

#define N_NODES  200000
#define N_EDGES  2000000
#define N_GRAPHS 10000
#define HID      96      // 75 padded to 3*32 for f16 WMMA K-chunks
#define BN_EPS   1e-5f

// ---- weight prep: transpose + pad + fp16 convert (column-major for B-frags) ----
__global__ void prep_weights(const float* __restrict__ w1, const float* __restrict__ b1,
                             const float* __restrict__ w2,
                             _Float16* __restrict__ w1t, _Float16* __restrict__ w2t,
                             float* __restrict__ b1p) {
  // w1: [2][32][75] -> w1t: [2][96][32]  (n-major; n>=75 zero)
  for (int idx = threadIdx.x; idx < 2 * HID * 32; idx += blockDim.x) {
    int l = idx / (HID * 32), r = idx % (HID * 32);
    int n = r / 32, k = r % 32;
    float v = (n < 75) ? w1[l * 32 * 75 + k * 75 + n] : 0.f;
    w1t[idx] = (_Float16)v;
  }
  // w2: [2][75][32] -> w2t: [2][32][96]  (n-major; k>=75 zero)
  for (int idx = threadIdx.x; idx < 2 * 32 * HID; idx += blockDim.x) {
    int l = idx / (32 * HID), r = idx % (32 * HID);
    int n = r / HID, k = r % HID;
    float v = (k < 75) ? w2[l * 75 * 32 + k * 32 + n] : 0.f;
    w2t[idx] = (_Float16)v;
  }
  for (int idx = threadIdx.x; idx < 2 * HID; idx += blockDim.x) {
    int l = idx / HID, j = idx % HID;
    b1p[idx] = (j < 75) ? b1[l * 75 + j] : 0.f;
  }
}

// ---- node embedding: h = x @ node_w + node_b  (14x32, VALU is fine) ----
__global__ void node_embed(const float* __restrict__ x, const float* __restrict__ nw,
                           const float* __restrict__ nb, float* __restrict__ h) {
  int tid = blockIdx.x * blockDim.x + threadIdx.x;
  int i = tid >> 5, c = tid & 31;
  if (i >= N_NODES) return;
  float acc = nb[c];
#pragma unroll
  for (int k = 0; k < 14; ++k) acc += x[i * 14 + k] * nw[k * 32 + c];
  h[i * 32 + c] = acc;
}

__global__ void zero_f32(float* __restrict__ p, int n) {
  int tid = blockIdx.x * blockDim.x + threadIdx.x;
  if (tid < n) p[tid] = 0.f;
}

// ---- msg = relu(h[src] + edge_embed) scattered into agg[dst] ----
__global__ void msg_scatter(const float* __restrict__ h, const int* __restrict__ ei,
                            const float* __restrict__ ea, const float* __restrict__ ew,
                            const float* __restrict__ eb, float* __restrict__ agg) {
  int tid = blockIdx.x * blockDim.x + threadIdx.x;
  int e = tid >> 5, c = tid & 31;
  if (e >= N_EDGES) return;
  int src = ei[e], dst = ei[N_EDGES + e];
  float a0 = ea[e * 3 + 0], a1 = ea[e * 3 + 1], a2 = ea[e * 3 + 2];
  float ec = eb[c] + a0 * ew[c] + a1 * ew[32 + c] + a2 * ew[64 + c];
  float m = h[src * 32 + c] + ec;
  if (m > 0.f) atomicAdd(&agg[dst * 32 + c], m);  // relu; adding 0 is a no-op
}

// ---- zin_f16 = (f16)(h + agg) : WMMA A-matrix operand buffer ----
__global__ void add_cvt(const float* __restrict__ h, const float* __restrict__ agg,
                        _Float16* __restrict__ zin) {
  int tid = blockIdx.x * blockDim.x + threadIdx.x;
  if (tid < N_NODES * 32) zin[tid] = (_Float16)(h[tid] + agg[tid]);
}

// ---- MLP (32->96->32) via WMMA f16, fused BatchNorm partial sums ----
__global__ __launch_bounds__(256) void mlp_wmma(
    const _Float16* __restrict__ zin, const _Float16* __restrict__ w1t,
    const float* __restrict__ b1p, const _Float16* __restrict__ w2t,
    const float* __restrict__ b2, float* __restrict__ z,
    float* __restrict__ bnsum, float* __restrict__ bnsumsq, int ntile) {
  __shared__ _Float16 lds[8][16 * HID];     // per-wave private hidden tile
  int wave = threadIdx.x >> 5;
  int lane = threadIdx.x & 31;
  int tile = blockIdx.x * 8 + wave;
  if (tile >= ntile) return;                // wave-uniform: EXEC stays all-1s
  int row0 = tile * 16;
  int m  = lane & 15;                       // A: row | B/C/D: column
  int kh = lane >> 4;                       // K-half / M-half select
  int mb = kh * 8;                          // C/D row base

  // A-frag: zin tile rows row0..row0+15, K=0..31 (16-bit A layout, ISA 7.12.2)
  v16h a;
  {
    const _Float16* p = zin + (row0 + m) * 32 + kh * 8;
#pragma unroll
    for (int i = 0; i < 8; ++i) { a[i] = p[i]; a[8 + i] = p[16 + i]; }
  }

  _Float16* hl = lds[wave];

  // GEMM1: hid[16x96] = relu(A @ w1t + b1), 6 N-tiles, single K-chunk
#pragma unroll
  for (int nt = 0; nt < 6; ++nt) {
    v16h b;
    {
      const _Float16* p = w1t + (nt * 16 + m) * 32 + kh * 8;  // col-major B
#pragma unroll
      for (int i = 0; i < 8; ++i) { b[i] = p[i]; b[8 + i] = p[16 + i]; }
    }
    v8f c;
    float bv = b1p[nt * 16 + m];
#pragma unroll
    for (int r = 0; r < 8; ++r) c[r] = bv;
    c = __builtin_amdgcn_wmma_f32_16x16x32_f16(false, a, false, b, (short)0, c,
                                               false, false);
#pragma unroll
    for (int r = 0; r < 8; ++r) {
      float v = c[r] > 0.f ? c[r] : 0.f;
      hl[(mb + r) * HID + nt * 16 + m] = (_Float16)v;
    }
  }

  // GEMM2: out[16x32] = hid @ w2t + b2, 2 N-tiles x 3 K-chunks
#pragma unroll
  for (int nt = 0; nt < 2; ++nt) {
    v8f c;
    float bv = b2[nt * 16 + m];
#pragma unroll
    for (int r = 0; r < 8; ++r) c[r] = bv;
#pragma unroll
    for (int kc = 0; kc < 3; ++kc) {
      v16h a2;
      {
        const _Float16* p = hl + m * HID + kc * 32 + kh * 8;
#pragma unroll
        for (int i = 0; i < 8; ++i) { a2[i] = p[i]; a2[8 + i] = p[16 + i]; }
      }
      v16h b;
      {
        const _Float16* p = w2t + (nt * 16 + m) * HID + kc * 32 + kh * 8;
#pragma unroll
        for (int i = 0; i < 8; ++i) { b[i] = p[i]; b[8 + i] = p[16 + i]; }
      }
      c = __builtin_amdgcn_wmma_f32_16x16x32_f16(false, a2, false, b, (short)0, c,
                                                 false, false);
    }
    float s = 0.f, s2 = 0.f;
#pragma unroll
    for (int r = 0; r < 8; ++r) {
      float v = c[r];
      z[(row0 + mb + r) * 32 + nt * 16 + m] = v;
      s += v; s2 += v * v;
    }
    // fold the two half-column lanes (n and n+16), then one atomic per column
    s  += __shfl_xor(s, 16, 32);
    s2 += __shfl_xor(s2, 16, 32);
    if (kh == 0) {
      atomicAdd(&bnsum[nt * 16 + m], s);
      atomicAdd(&bnsumsq[nt * 16 + m], s2);
    }
  }
}

__global__ void bn_finalize(const float* __restrict__ s, const float* __restrict__ s2,
                            const float* __restrict__ g, const float* __restrict__ b,
                            float* __restrict__ scale, float* __restrict__ shift) {
  int c = threadIdx.x;
  if (c >= 32) return;
  float inv_n = 1.f / (float)N_NODES;
  float mu = s[c] * inv_n;
  float var = fmaxf(s2[c] * inv_n - mu * mu, 0.f);
  float sc = g[c] * rsqrtf(var + BN_EPS);
  scale[c] = sc;
  shift[c] = b[c] - mu * sc;
}

__global__ void bn_apply(const float* __restrict__ z, const float* __restrict__ scale,
                         const float* __restrict__ shift, float* __restrict__ h) {
  int tid = blockIdx.x * blockDim.x + threadIdx.x;
  if (tid >= N_NODES * 32) return;
  int c = tid & 31;
  float v = z[tid] * scale[c] + shift[c];
  h[tid] = v > 0.f ? v : 0.f;
}

__global__ void pool(const float* __restrict__ h, const int* __restrict__ batch,
                     float* __restrict__ gsum, float* __restrict__ gcnt) {
  int tid = blockIdx.x * blockDim.x + threadIdx.x;
  int i = tid >> 5, c = tid & 31;
  if (i >= N_NODES) return;
  int b = batch[i];
  atomicAdd(&gsum[b * 32 + c], h[i * 32 + c]);
  if (c == 0) atomicAdd(&gcnt[b], 1.f);
}

__global__ void head(const float* __restrict__ gsum, const float* __restrict__ gcnt,
                     const float* __restrict__ w1, const float* __restrict__ b1,
                     const float* __restrict__ w2, const float* __restrict__ b2,
                     float* __restrict__ out) {
  int g = blockIdx.x * blockDim.x + threadIdx.x;
  if (g >= N_GRAPHS) return;
  float inv = 1.f / fmaxf(gcnt[g], 1.f);
  float v[32];
#pragma unroll
  for (int c = 0; c < 32; ++c) v[c] = gsum[g * 32 + c] * inv;
  float t[16];
#pragma unroll
  for (int j = 0; j < 16; ++j) {
    float acc = b1[j];
#pragma unroll
    for (int c = 0; c < 32; ++c) acc += v[c] * w1[c * 16 + j];
    t[j] = acc > 0.f ? acc : 0.f;
  }
#pragma unroll
  for (int o = 0; o < 2; ++o) {
    float acc = b2[o];
#pragma unroll
    for (int j = 0; j < 16; ++j) acc += t[j] * w2[j * 2 + o];
    out[g * 2 + o] = acc;
  }
}

extern "C" void kernel_launch(void* const* d_in, const int* in_sizes, int n_in,
                              void* d_out, int out_size, void* d_ws, size_t ws_size,
                              hipStream_t stream) {
  (void)in_sizes; (void)n_in; (void)out_size; (void)ws_size;
  const float* x      = (const float*)d_in[0];
  const int*   ei     = (const int*)d_in[1];
  const float* eattr  = (const float*)d_in[2];
  const int*   batch  = (const int*)d_in[3];
  const float* node_w = (const float*)d_in[4];
  const float* node_b = (const float*)d_in[5];
  const float* edge_w = (const float*)d_in[6];
  const float* edge_b = (const float*)d_in[7];
  const float* w1  = (const float*)d_in[8];
  const float* b1  = (const float*)d_in[9];
  const float* w2  = (const float*)d_in[10];
  const float* b2  = (const float*)d_in[11];
  const float* bng = (const float*)d_in[12];
  const float* bnb = (const float*)d_in[13];
  const float* l1w = (const float*)d_in[14];
  const float* l1b = (const float*)d_in[15];
  const float* l2w = (const float*)d_in[16];
  const float* l2b = (const float*)d_in[17];
  float* out = (float*)d_out;

  // workspace carve-out (~91 MB), 256B aligned slices
  char* wsb = (char*)d_ws;
  size_t off = 0;
  auto take = [&](size_t bytes) -> char* {
    char* p = wsb + off;
    off = (off + bytes + 255) & ~(size_t)255;
    return p;
  };
  float*    h    = (float*)take((size_t)N_NODES * 32 * sizeof(float));
  float*    agg  = (float*)take((size_t)N_NODES * 32 * sizeof(float));
  float*    z    = (float*)take((size_t)N_NODES * 32 * sizeof(float));
  _Float16* zin  = (_Float16*)take((size_t)N_NODES * 32 * sizeof(_Float16));
  float*    gsum = (float*)take((size_t)N_GRAPHS * 33 * sizeof(float)); // gsum + gcnt
  float*    gcnt = gsum + (size_t)N_GRAPHS * 32;
  float*    stats = (float*)take(128 * sizeof(float));
  float *bnsum = stats, *bnsumsq = stats + 32, *scale = stats + 64, *shift = stats + 96;
  _Float16* w1t = (_Float16*)take((size_t)2 * HID * 32 * sizeof(_Float16));
  _Float16* w2t = (_Float16*)take((size_t)2 * 32 * HID * sizeof(_Float16));
  float*    b1p = (float*)take((size_t)2 * HID * sizeof(float));

  const int n32 = N_NODES * 32;               // 6,400,000
  prep_weights<<<1, 256, 0, stream>>>(w1, b1, w2, w1t, w2t, b1p);
  node_embed<<<n32 / 256, 256, 0, stream>>>(x, node_w, node_b, h);

  const int ntile = N_NODES / 16;             // 12500
  for (int l = 0; l < 2; ++l) {
    zero_f32<<<(n32 + 255) / 256, 256, 0, stream>>>(agg, n32);
    zero_f32<<<1, 64, 0, stream>>>(bnsum, 64);
    msg_scatter<<<(N_EDGES * 32) / 256, 256, 0, stream>>>(h, ei, eattr, edge_w,
                                                          edge_b, agg);
    add_cvt<<<n32 / 256, 256, 0, stream>>>(h, agg, zin);
    mlp_wmma<<<(ntile + 7) / 8, 256, 0, stream>>>(
        zin, w1t + (size_t)l * HID * 32, b1p + (size_t)l * HID,
        w2t + (size_t)l * 32 * HID, b2 + (size_t)l * 32, z, bnsum, bnsumsq, ntile);
    bn_finalize<<<1, 32, 0, stream>>>(bnsum, bnsumsq, bng + l * 32, bnb + l * 32,
                                      scale, shift);
    bn_apply<<<n32 / 256, 256, 0, stream>>>(z, scale, shift, h);
  }

  zero_f32<<<(N_GRAPHS * 33 + 255) / 256, 256, 0, stream>>>(gsum, N_GRAPHS * 33);
  pool<<<n32 / 256, 256, 0, stream>>>(h, batch, gsum, gcnt);
  head<<<(N_GRAPHS + 255) / 256, 256, 0, stream>>>(gsum, gcnt, l1w, l1b, l2w, l2b,
                                                   out);
}
